// GIN_51427938402588
// MI455X (gfx1250) — compile-verified, matrix operand order checked by reading, not run on hardware
//
#include <hip/hip_runtime.h>

typedef __attribute__((ext_vector_type(16))) __bf16 v16bf;
typedef __attribute__((ext_vector_type(8)))  __bf16 v8bf;
typedef __attribute__((ext_vector_type(8)))  float  v8f;

// ---------------- aggregation ----------------

__global__ void k_copy(const float* __restrict__ src, float* __restrict__ dst, int n4) {
    int i = blockIdx.x * blockDim.x + threadIdx.x;
    if (i < n4) ((float4*)dst)[i] = ((const float4*)src)[i];
}

__global__ void k_scatter(const float* __restrict__ x, const int* __restrict__ srcI,
                          const int* __restrict__ dstI, float* __restrict__ agg,
                          int E, int Kc, int K) {
    int i = blockIdx.x * blockDim.x + threadIdx.x;
    if (i >= E * Kc) return;
    int e = i / Kc, c = i % Kc;
    int s = srcI[e], d = dstI[e];
    float4 v = ((const float4*)(x + (size_t)s * K))[c];
    float* p = agg + (size_t)d * K + c * 4;
    unsafeAtomicAdd(p + 0, v.x);
    unsafeAtomicAdd(p + 1, v.y);
    unsafeAtomicAdd(p + 2, v.z);
    unsafeAtomicAdd(p + 3, v.w);
}

__global__ void k_zero(float* __restrict__ p, int n) {
    int i = blockIdx.x * blockDim.x + threadIdx.x;
    if (i < n) p[i] = 0.0f;
}

// Stage W (f32, KxN row-major) into LDS pre-swizzled into WMMA B-fragment
// layout: Wswz[(nt*KT + kt)*32 + lane][e], so each lane reads its 16 bf16
// contiguously (2x ds_load_b128) instead of 16x ds_load_u16.
template <int K, int N>
__device__ __forceinline__ void stage_wswz(const float* __restrict__ W, __bf16* Wswz, int tid) {
    constexpr int KT = K / 32;
    for (int idx = tid; idx < K * N; idx += 256) {
        int e  = idx & 15;
        int ln = (idx >> 4) & 31;
        int fk = idx >> 9;          // fragment id = nt*KT + kt
        int kt = fk % KT;
        int nt = fk / KT;
        int kk  = kt * 32 + ((ln >> 4) * 8) + (e < 8 ? e : e + 8);
        int col = nt * 16 + (ln & 15);
        Wswz[idx] = (__bf16)W[kk * N + col];
    }
}

// ---------------- GEMM1: h = agg @ W1 + b1, emit bf16 h + BN stats ----------------

template <int K, int N>
__global__ void __launch_bounds__(256)
k_gemm1(const float* __restrict__ A, const float* __restrict__ W,
        const float* __restrict__ bias, __bf16* __restrict__ H,
        float* __restrict__ stats, int M) {
    constexpr int KT = K / 32;
    constexpr int NT = N / 16;
    __shared__ __bf16 Wswz[K * N];
    int tid = threadIdx.x;
    stage_wswz<K, N>(W, Wswz, tid);
    __syncthreads();

    int lane = tid & 31, wave = tid >> 5;
    int mt0 = (blockIdx.x * 8 + wave) * 2;   // two 16-row tiles per wave
    if (mt0 * 16 >= M) return;
    bool has1 = ((mt0 + 1) * 16) < M;        // wave-uniform

    int r0 = mt0 * 16 + (lane & 15);
    int r1 = (mt0 + 1) * 16 + (lane & 15);
    if (r1 >= M) r1 = M - 1;
    int koff  = (lane >> 4) * 8;
    int col0  = lane & 15;
    int rbase = (lane >> 4) * 8;

    v16bf a0[KT], a1[KT];
#pragma unroll
    for (int kt = 0; kt < KT; ++kt) {
        {
            const float* p = A + (size_t)r0 * K + kt * 32 + koff;
            float t[16];
            *(float4*)(t + 0)  = *(const float4*)(p + 0);
            *(float4*)(t + 4)  = *(const float4*)(p + 4);
            *(float4*)(t + 8)  = *(const float4*)(p + 16);
            *(float4*)(t + 12) = *(const float4*)(p + 20);
#pragma unroll
            for (int e = 0; e < 16; ++e) a0[kt][e] = (__bf16)t[e];
        }
        {
            const float* p = A + (size_t)r1 * K + kt * 32 + koff;
            float t[16];
            *(float4*)(t + 0)  = *(const float4*)(p + 0);
            *(float4*)(t + 4)  = *(const float4*)(p + 4);
            *(float4*)(t + 8)  = *(const float4*)(p + 16);
            *(float4*)(t + 12) = *(const float4*)(p + 20);
#pragma unroll
            for (int e = 0; e < 16; ++e) a1[kt][e] = (__bf16)t[e];
        }
    }

    const v16bf* frg = (const v16bf*)Wswz;
#pragma unroll
    for (int nt = 0; nt < NT; ++nt) {
        int col = nt * 16 + col0;
        v8f acc0 = {}, acc1 = {};
#pragma unroll
        for (int kt = 0; kt < KT; ++kt) {
            v16bf b = frg[(nt * KT + kt) * 32 + lane];
            acc0 = __builtin_amdgcn_wmma_f32_16x16x32_bf16(
                false, a0[kt], false, b, (short)0, acc0, false, false);
            acc1 = __builtin_amdgcn_wmma_f32_16x16x32_bf16(
                false, a1[kt], false, b, (short)0, acc1, false, false);
        }
        float bc = bias[col];
        float s = 0.0f, sq = 0.0f;
#pragma unroll
        for (int r = 0; r < 8; ++r) {
            float v = acc0[r] + bc;
            H[(size_t)(mt0 * 16 + rbase + r) * N + col] = (__bf16)v;
            s += v;
            sq += v * v;
        }
        if (has1) {
#pragma unroll
            for (int r = 0; r < 8; ++r) {
                float v = acc1[r] + bc;
                H[(size_t)((mt0 + 1) * 16 + rbase + r) * N + col] = (__bf16)v;
                s += v;
                sq += v * v;
            }
        }
        s  += __shfl_xor(s, 16, 32);
        sq += __shfl_xor(sq, 16, 32);
        if (lane < 16) {
            unsafeAtomicAdd(&stats[col], s);
            unsafeAtomicAdd(&stats[N + col], sq);
        }
    }
}

// ---------------- BN finalize: scale/shift per column ----------------

__global__ void k_bnfin(const float* __restrict__ stats, const float* __restrict__ g,
                        const float* __restrict__ be, float* __restrict__ ss,
                        int N, float invM) {
    int n = blockIdx.x * blockDim.x + threadIdx.x;
    if (n >= N) return;
    float mu  = stats[n] * invM;
    float var = stats[N + n] * invM - mu * mu;
    float inv = rsqrtf(var + 1e-5f);
    float sc  = g[n] * inv;
    ss[n]     = sc;
    ss[N + n] = be[n] - mu * sc;
}

// ---------------- GEMM2: out = relu(h*scale+shift) @ W2 + b2 ----------------

template <int K, int N>
__global__ void __launch_bounds__(256)
k_gemm2(const __bf16* __restrict__ H, const float* __restrict__ ss,
        const float* __restrict__ W, const float* __restrict__ bias,
        float* __restrict__ Out, int M) {
    constexpr int KT = K / 32;
    constexpr int NT = N / 16;
    __shared__ __bf16 Wswz[K * N];
    __shared__ float scale_s[K];
    __shared__ float shift_s[K];
    int tid = threadIdx.x;
    stage_wswz<K, N>(W, Wswz, tid);
    for (int idx = tid; idx < K; idx += 256) {
        scale_s[idx] = ss[idx];
        shift_s[idx] = ss[K + idx];
    }
    __syncthreads();

    int lane = tid & 31, wave = tid >> 5;
    int mt0 = (blockIdx.x * 8 + wave) * 2;
    if (mt0 * 16 >= M) return;
    bool has1 = ((mt0 + 1) * 16) < M;

    int r0 = mt0 * 16 + (lane & 15);
    int r1 = (mt0 + 1) * 16 + (lane & 15);
    if (r1 >= M) r1 = M - 1;
    int koff  = (lane >> 4) * 8;
    int col0  = lane & 15;
    int rbase = (lane >> 4) * 8;

    v16bf a0[KT], a1[KT];
#pragma unroll
    for (int kt = 0; kt < KT; ++kt) {
        {
            const __bf16* p = H + (size_t)r0 * K + kt * 32 + koff;
            v8bf h0 = *(const v8bf*)(p);
            v8bf h1 = *(const v8bf*)(p + 16);
#pragma unroll
            for (int e = 0; e < 8; ++e) {
                int k0 = kt * 32 + koff + e;
                int k1 = k0 + 16;
                float v0 = (float)h0[e] * scale_s[k0] + shift_s[k0];
                float v1 = (float)h1[e] * scale_s[k1] + shift_s[k1];
                a0[kt][e]     = (__bf16)(v0 > 0.0f ? v0 : 0.0f);
                a0[kt][e + 8] = (__bf16)(v1 > 0.0f ? v1 : 0.0f);
            }
        }
        {
            const __bf16* p = H + (size_t)r1 * K + kt * 32 + koff;
            v8bf h0 = *(const v8bf*)(p);
            v8bf h1 = *(const v8bf*)(p + 16);
#pragma unroll
            for (int e = 0; e < 8; ++e) {
                int k0 = kt * 32 + koff + e;
                int k1 = k0 + 16;
                float v0 = (float)h0[e] * scale_s[k0] + shift_s[k0];
                float v1 = (float)h1[e] * scale_s[k1] + shift_s[k1];
                a1[kt][e]     = (__bf16)(v0 > 0.0f ? v0 : 0.0f);
                a1[kt][e + 8] = (__bf16)(v1 > 0.0f ? v1 : 0.0f);
            }
        }
    }

    const v16bf* frg = (const v16bf*)Wswz;
#pragma unroll
    for (int nt = 0; nt < NT; ++nt) {
        int col = nt * 16 + col0;
        v8f acc0 = {}, acc1 = {};
#pragma unroll
        for (int kt = 0; kt < KT; ++kt) {
            v16bf b = frg[(nt * KT + kt) * 32 + lane];
            acc0 = __builtin_amdgcn_wmma_f32_16x16x32_bf16(
                false, a0[kt], false, b, (short)0, acc0, false, false);
            acc1 = __builtin_amdgcn_wmma_f32_16x16x32_bf16(
                false, a1[kt], false, b, (short)0, acc1, false, false);
        }
        float bc = bias[col];
#pragma unroll
        for (int r = 0; r < 8; ++r) {
            Out[(size_t)(mt0 * 16 + rbase + r) * N + col] = acc0[r] + bc;
        }
        if (has1) {
#pragma unroll
            for (int r = 0; r < 8; ++r) {
                Out[(size_t)((mt0 + 1) * 16 + rbase + r) * N + col] = acc1[r] + bc;
            }
        }
    }
}

// ---------------- host orchestration ----------------

extern "C" void kernel_launch(void* const* d_in, const int* in_sizes, int n_in,
                              void* d_out, int out_size, void* d_ws, size_t ws_size,
                              hipStream_t stream) {
    const int M = in_sizes[0] / 64;      // 100000 nodes
    const int E = in_sizes[1] / 2;       // 1e6 edges
    const float* x = (const float*)d_in[0];
    const int* srcI = (const int*)d_in[1];
    const int* dstI = srcI + E;

    const float* W1[3] = {(const float*)d_in[2],  (const float*)d_in[8],  (const float*)d_in[14]};
    const float* B1[3] = {(const float*)d_in[3],  (const float*)d_in[9],  (const float*)d_in[15]};
    const float* Ga[3] = {(const float*)d_in[4],  (const float*)d_in[10], (const float*)d_in[16]};
    const float* Be[3] = {(const float*)d_in[5],  (const float*)d_in[11], (const float*)d_in[17]};
    const float* W2[3] = {(const float*)d_in[6],  (const float*)d_in[12], (const float*)d_in[18]};
    const float* B2[3] = {(const float*)d_in[7],  (const float*)d_in[13], (const float*)d_in[19]};

    const int Kin[3]  = {64, 128, 128};  // GEMM1 K
    const int Dmid[3] = {128, 128, 64};  // GEMM1 N == GEMM2 K == GEMM2 N

    char* ws = (char*)d_ws;
    float*  agg   = (float*)ws;                                               // M*128 f32
    __bf16* hbuf  = (__bf16*)(ws + (size_t)M * 128 * 4);                      // M*128 bf16
    float*  ybuf  = (float*)(ws + (size_t)M * 128 * 4 + (size_t)M * 128 * 2); // M*128 f32
    float*  stats = (float*)(ws + (size_t)M * 128 * 4 + (size_t)M * 128 * 2 +
                             (size_t)M * 128 * 4);                            // 256 f32
    float*  ssbuf = stats + 256;                                              // 256 f32

    const float* inPtr = x;
    for (int l = 0; l < 3; ++l) {
        const int K = Kin[l], D = Dmid[l];

        // agg = x + segment_sum(x[src] -> dst)
        int n4 = M * K / 4;
        k_copy<<<(n4 + 255) / 256, 256, 0, stream>>>(inPtr, agg, n4);
        int tot = E * (K / 4);
        k_scatter<<<(tot + 255) / 256, 256, 0, stream>>>(inPtr, srcI, dstI, agg, E, K / 4, K);

        k_zero<<<1, 256, 0, stream>>>(stats, 2 * D);

        int tiles = (M + 15) / 16;
        int nblk  = (tiles + 15) / 16;   // 8 waves/block, 2 tiles/wave
        if (l == 0)      k_gemm1<64, 128><<<nblk, 256, 0, stream>>>(agg, W1[l], B1[l], hbuf, stats, M);
        else if (l == 1) k_gemm1<128, 128><<<nblk, 256, 0, stream>>>(agg, W1[l], B1[l], hbuf, stats, M);
        else             k_gemm1<128, 64><<<nblk, 256, 0, stream>>>(agg, W1[l], B1[l], hbuf, stats, M);

        k_bnfin<<<1, 128, 0, stream>>>(stats, Ga[l], Be[l], ssbuf, D, 1.0f / (float)M);

        float* outP = (l == 2) ? (float*)d_out : ybuf;
        if (l == 2) k_gemm2<64, 64><<<nblk, 256, 0, stream>>>(hbuf, ssbuf, W2[l], B2[l], outP, M);
        else        k_gemm2<128, 128><<<nblk, 256, 0, stream>>>(hbuf, ssbuf, W2[l], B2[l], outP, M);

        inPtr = ybuf;
    }
}
